// MultiHeadAttention_9397388443950
// MI455X (gfx1250) — matrix operand order repaired and checked by reading, not compiled, not run on hardware
//
#include <hip/hip_runtime.h>
#include <hip/hip_bf16.h>
#include <stdint.h>

// ---------------------------------------------------------------------------
// MHA w/ RoPE + KV cache for MI455X (gfx1250), wave32, WMMA bf16 path.
// D = A x B + C with V_WMMA_F32_16X16X32_BF16 for all matmul stages.
// GEMM stages its shared B tile through LDS via async global->LDS copies
// (ASYNCcnt / s_wait_asynccnt) when the toolchain exposes the builtin.
// ---------------------------------------------------------------------------

typedef __bf16 bf16_t;
typedef __attribute__((ext_vector_type(16))) __bf16 v16bf;
typedef __attribute__((ext_vector_type(8)))  __bf16 v8bf;
typedef __attribute__((ext_vector_type(8)))  float  v8f;
typedef __attribute__((ext_vector_type(4)))  int    v4i;

typedef __attribute__((address_space(1))) v4i gv4i;   // global int4
typedef __attribute__((address_space(3))) v4i lv4i;   // LDS int4

#if defined(__has_builtin)
# if __has_builtin(__builtin_amdgcn_global_load_async_to_lds_b128)
#  define HAVE_ASYNC_LDS 1
# endif
#endif
#ifndef HAVE_ASYNC_LDS
# define HAVE_ASYNC_LDS 0
#endif

constexpr int   Bc  = 2;
constexpr int   Sc  = 1024;
constexpr int   SPc = 1024;
constexpr int   Ec  = 2048;
constexpr int   E3  = 3 * Ec;      // 6144
constexpr int   Hc  = 16;
constexpr int   HDc = 128;
constexpr int   LT  = SPc + Sc;    // 2048 total keys
constexpr float BASEF  = 10000.0f;
constexpr float SCALEF = 0.08838834764831845f;   // 1/sqrt(128)

// ---------------- helpers ----------------

static __device__ inline bf16_t f2bf(float x) {
  unsigned int u = __float_as_uint(x);
  u += 0x7FFFu + ((u >> 16) & 1u);               // round-to-nearest-even
  unsigned short h = (unsigned short)(u >> 16);
  return __builtin_bit_cast(bf16_t, h);
}

static __device__ inline v16bf cat8(v8bf lo, v8bf hi) {
  return __builtin_shufflevector(lo, hi, 0,1,2,3,4,5,6,7,8,9,10,11,12,13,14,15);
}

static __device__ inline void stage_wait() {
#if HAVE_ASYNC_LDS
  asm volatile("s_wait_asynccnt 0" ::: "memory");
#else
  asm volatile("s_wait_dscnt 0" ::: "memory");
#endif
}

// A-operand fragment: 16x32 bf16 tile from row-major src (ld elements / row).
// lane L: row = row0 + (L&15); halves = 16 contiguous K values kb..kb+15,
// kb = 16*(L>=16).   (contiguous k order, not the ISA table order; B fragments match)
static __device__ inline v16bf load_frag_a(const bf16_t* __restrict__ src,
                                           int ld, int row0, int k0) {
  const int lane = threadIdx.x & 31;
  const int r    = lane & 15;
  const int kb   = (lane >> 4) << 4;             // 0 or 16
  const bf16_t* p = src + (size_t)(row0 + r) * ld + k0 + kb;
  v8bf lo = *(const v8bf*)(p);
  v8bf hi = *(const v8bf*)(p + 8);
  return cat8(lo, hi);
}

// B-operand fragment: computes with the 32x16 (KxN) matrix whose column n is
// row (n0+n) of the row-major src -> i.e. contributes src-row dot products
// (A @ src^T).  lane L: col = n0 + (L&15); halves = 16 contiguous K values at
// k0 + 16*(L>=16).   (contiguous k order, same as the A fragment)
static __device__ inline v16bf load_frag_b(const bf16_t* __restrict__ src,
                                           int ld, int n0, int k0) {
  const int lane = threadIdx.x & 31;
  const int n    = lane & 15;
  const int kb   = (lane >> 4) << 4;             // 0 or 16
  const bf16_t* p = src + (size_t)(n0 + n) * ld + k0 + kb;
  v8bf lo = *(const v8bf*)(p);
  v8bf hi = *(const v8bf*)(p + 8);
  return cat8(lo, hi);
}

static __device__ inline v8f wmma_bf16(v16bf a, v16bf b, v8f c) {
  return __builtin_amdgcn_wmma_f32_16x16x32_bf16(false, a, false, b,
                                                 (short)0, c, false, false);
}

// ---------------- fp32 -> bf16 convert (8 elems / thread) ----------------

__global__ void cvt_bf16_kernel(const float* __restrict__ src,
                                bf16_t* __restrict__ dst, int n8) {
  int i = blockIdx.x * blockDim.x + threadIdx.x;
  if (i >= n8) return;
  const float4* s = (const float4*)src + (size_t)i * 2;
  float4 a = s[0], b = s[1];
  v8bf o;
  o[0]=f2bf(a.x); o[1]=f2bf(a.y); o[2]=f2bf(a.z); o[3]=f2bf(a.w);
  o[4]=f2bf(b.x); o[5]=f2bf(b.y); o[6]=f2bf(b.z); o[7]=f2bf(b.w);
  *((v8bf*)dst + i) = o;
}

// ---------------- generic C(f32) = A(bf16,MxK) @ B(bf16,NxK)^T -------------
// 256 thr = 8 waves; block tile 128(M) x 64(N); wave w -> 16-row strip,
// 4 accumulators of 16x16.  The 64x32 B tile for each k-step is staged in
// LDS (async global->LDS copy, double buffered) and shared by all 8 waves;
// A fragments stream straight from global (unique rows per wave).

__global__ void __launch_bounds__(256)
gemm_bf16_kernel(const bf16_t* __restrict__ A, const bf16_t* __restrict__ Bm,
                 float* __restrict__ C, int M, int N, int K) {
  __shared__ bf16_t btile[2][64][32];              // 2 x 4 KB

  const int w    = threadIdx.x >> 5;
  const int lane = threadIdx.x & 31;
  const int row0 = blockIdx.y * 128 + w * 16;
  const int col0 = blockIdx.x * 64;

  // staging: thread t moves one 16B chunk of the 64x32 bf16 B tile
  const int srow = threadIdx.x >> 2;               // 0..63
  const int scol = (threadIdx.x & 3) * 8;          // 0,8,16,24

  auto stage = [&](int k0, int buf) {
    const bf16_t* g = Bm + (size_t)(col0 + srow) * K + k0 + scol;
    bf16_t*       l = &btile[buf][srow][scol];
#if HAVE_ASYNC_LDS
    __builtin_amdgcn_global_load_async_to_lds_b128((gv4i*)g, (lv4i*)l, 0, 0);
#else
    *(v8bf*)l = *(const v8bf*)g;
#endif
  };

  const int nsteps = K / 32;
  stage(0, 0);

  v8f acc[4] = {};
  for (int s = 0; s < nsteps; ++s) {
    const int k0  = s * 32;
    const int buf = s & 1;
    stage_wait();            // our stage for this step has landed in LDS
    __syncthreads();         // every wave's stage has landed; prev reads done
    if (s + 1 < nsteps) stage(k0 + 32, buf ^ 1);   // overlap copy with WMMAs

    // speculative prefetch of the next A chunk (clamped, branch-free)
    int kpf = k0 + 32 < K ? k0 + 32 : k0;
    __builtin_prefetch(A + (size_t)(row0 + (lane & 15)) * K + kpf, 0, 1);

    v16bf af = load_frag_a(A, K, row0, k0);
#pragma unroll
    for (int j = 0; j < 4; ++j) {
      const v8bf* p =
          (const v8bf*)&btile[buf][j * 16 + (lane & 15)][(lane >> 4) * 16];
      v16bf bfr = cat8(p[0], p[1]);
      acc[j] = wmma_bf16(af, bfr, acc[j]);
    }
  }

  const int n16 = lane & 15;
  const int mb  = (lane >> 4) << 3;
#pragma unroll
  for (int j = 0; j < 4; ++j)
#pragma unroll
    for (int i = 0; i < 8; ++i)
      C[(size_t)(row0 + mb + i) * N + col0 + j * 16 + n16] = acc[j][i];
}

// ---------------- KV cache repack: past K/V (f32 [b,sp,h,d]) ---------------
// -> Kb bf16 [b,h,L,d] rows; Vt bf16 [b,h,d,L] (V transposed).

__global__ void cache_copy_kernel(const float* __restrict__ pk,
                                  const float* __restrict__ pv,
                                  bf16_t* __restrict__ Kb,
                                  bf16_t* __restrict__ Vt) {
  int tid = blockIdx.x * blockDim.x + threadIdx.x;
  if (tid >= Bc * SPc * Hc * HDc) return;
  int d  = tid & (HDc - 1);
  int h  = (tid >> 7) & (Hc - 1);
  int sp = (tid >> 11) & (SPc - 1);
  int b  = tid >> 21;
  Kb[(((size_t)(b * Hc + h)) * LT + sp) * HDc + d]  = f2bf(pk[tid]);
  Vt[(((size_t)(b * Hc + h)) * HDc + d) * LT + sp]  = f2bf(pv[tid]);
}

// ---------------- RoPE on q,k + scatter of new k,v into caches -------------

__global__ void rope_scatter_kernel(const float* __restrict__ qkv,
                                    bf16_t* __restrict__ Qb,
                                    bf16_t* __restrict__ Kb,
                                    bf16_t* __restrict__ Vt) {
  int tid = blockIdx.x * blockDim.x + threadIdx.x;
  if (tid >= Bc * Sc * Hc * HDc) return;
  int d = tid & (HDc - 1);
  int h = (tid >> 7) & (Hc - 1);
  int s = (tid >> 11) & (Sc - 1);
  int b = tid >> 21;

  const float* row = qkv + (size_t)(b * Sc + s) * E3;
  int   col = h * HDc + d;
  int   dp  = d ^ 1;
  float q   = row[col];
  float k   = row[Ec + col];
  float v   = row[2 * Ec + col];
  float qp  = row[h * HDc + dp];
  float kp  = row[Ec + h * HDc + dp];
  // rotate_half: rh[2i] = -x[2i+1]; rh[2i+1] = x[2i]
  float rq = (d & 1) ? qp : -qp;
  float rk = (d & 1) ? kp : -kp;

  // emb[d] = t * BASE^(-2*(d%64)/HD)
  float t   = (float)(SPc + s);
  float ang = t * __powf(BASEF, -(float)((d & 63) * 2) / (float)HDc);
  float cs  = __cosf(ang), sn = __sinf(ang);

  Qb[(((size_t)(b * Hc + h)) * Sc + s) * HDc + d]         = f2bf(q * cs + rq * sn);
  Kb[(((size_t)(b * Hc + h)) * LT + (SPc + s)) * HDc + d] = f2bf(k * cs + rk * sn);
  Vt[(((size_t)(b * Hc + h)) * HDc + d) * LT + (SPc + s)] = f2bf(v);
}

// ---------------- flash-attention: 4 independent waves / block -------------
// grid = (B*H, S/64); wave w owns q rows [qstart, qstart+16).
// Per 32-key block: 8 WMMA (scores) + 8 WMMA (P.V); online softmax via LDS.

__global__ void __launch_bounds__(128)
attn_kernel(const bf16_t* __restrict__ Qb, const bf16_t* __restrict__ Kb,
            const bf16_t* __restrict__ Vt, bf16_t* __restrict__ attb) {
  __shared__ float  sc[4][16][32];
  __shared__ bf16_t pbuf[4][16][32];
  __shared__ float  rstate[4][16];

  const int w      = threadIdx.x >> 5;
  const int lane   = threadIdx.x & 31;
  const int n16    = lane & 15;
  const int mb     = (lane >> 4) << 3;
  const int bh     = blockIdx.x;                  // b*H + h
  const int qstart = blockIdx.y * 64 + w * 16;

  const bf16_t* Qh = Qb + (size_t)bh * Sc * HDc;
  const bf16_t* Kh = Kb + (size_t)bh * LT * HDc;
  const bf16_t* Vh = Vt + (size_t)bh * HDc * LT;

  v16bf qf[4];
#pragma unroll
  for (int c = 0; c < 4; ++c) qf[c] = load_frag_a(Qh, HDc, qstart, c * 32);

  v8f O[8] = {};
  float mrow = -1e30f, lrow = 0.0f;

  const int kend = (SPc + qstart + 16 + 31) & ~31;  // causal frontier, /32
  for (int kb = 0; kb < kend; kb += 32) {
    // ---- scores: 16x32 tile = Q(16x128) . K^T ----
    v8f s0 = {}, s1 = {};
#pragma unroll
    for (int c = 0; c < 4; ++c) {
      v16bf kf0 = load_frag_b(Kh, HDc, kb,      c * 32);
      v16bf kf1 = load_frag_b(Kh, HDc, kb + 16, c * 32);
      s0 = wmma_bf16(qf[c], kf0, s0);
      s1 = wmma_bf16(qf[c], kf1, s1);
    }
    // ---- scale + causal mask, stage in LDS (C layout -> row major) ----
#pragma unroll
    for (int j = 0; j < 8; ++j) {
      int qpos = SPc + qstart + mb + j;
      float v0 = (kb + n16      <= qpos) ? s0[j] * SCALEF : -1e30f;
      float v1 = (kb + 16 + n16 <= qpos) ? s1[j] * SCALEF : -1e30f;
      sc[w][mb + j][n16]      = v0;
      sc[w][mb + j][16 + n16] = v1;
    }
    asm volatile("s_wait_dscnt 0" ::: "memory");
    // ---- per-row online softmax update (lanes 0..15 own row r) ----
    if (lane < 16) {
      const int r = lane;
      float bm = -1e30f;
#pragma unroll
      for (int j = 0; j < 32; ++j) bm = fmaxf(bm, sc[w][r][j]);
      float mnew  = fmaxf(mrow, bm);
      float alpha = __expf(mrow - mnew);
      float lsum  = 0.0f;
#pragma unroll
      for (int j = 0; j < 32; ++j) {
        float p = __expf(sc[w][r][j] - mnew);
        lsum += p;
        pbuf[w][r][j] = f2bf(p);
      }
      lrow = lrow * alpha + lsum;
      mrow = mnew;
      rstate[w][r] = alpha;
    }
    asm volatile("s_wait_dscnt 0" ::: "memory");
    // ---- rescale accumulators by per-row alpha ----
#pragma unroll
    for (int j = 0; j < 8; ++j) {
      float a = rstate[w][mb + j];
#pragma unroll
      for (int t = 0; t < 8; ++t) O[t][j] *= a;
    }
    // ---- P fragment (A layout) from LDS, then O += P . V ----
    const v8bf* pr = (const v8bf*)&pbuf[w][n16][0];
    const int   ph = lane >> 4;                   // 0 or 1
    v16bf pf = cat8(pr[2 * ph], pr[2 * ph + 1]);
#pragma unroll
    for (int t = 0; t < 8; ++t) {
      v16bf vf = load_frag_b(Vh, LT, t * 16, kb); // rows of V^T = cols of V
      O[t] = wmma_bf16(pf, vf, O[t]);
    }
  }
  // ---- finalize: divide by l, emit bf16 rows of the attention output ----
  if (lane < 16) rstate[w][lane] = 1.0f / lrow;
  asm volatile("s_wait_dscnt 0" ::: "memory");
  const int b = bh >> 4, h = bh & 15;
#pragma unroll
  for (int j = 0; j < 8; ++j) {
    float  inv = rstate[w][mb + j];
    size_t row = (size_t)(b * Sc + qstart + mb + j) * Ec + h * HDc;
#pragma unroll
    for (int t = 0; t < 8; ++t)
      attb[row + t * 16 + n16] = f2bf(O[t][j] * inv);
  }
}

// ---------------------------------------------------------------------------

extern "C" void kernel_launch(void* const* d_in, const int* in_sizes, int n_in,
                              void* d_out, int out_size, void* d_ws, size_t ws_size,
                              hipStream_t stream) {
  (void)in_sizes; (void)n_in; (void)out_size; (void)ws_size;
  const float* x    = (const float*)d_in[0];
  const float* Wqkv = (const float*)d_in[1];
  const float* Wout = (const float*)d_in[2];
  const float* pk   = (const float*)d_in[3];
  const float* pv   = (const float*)d_in[4];
  // d_in[5] attn_mask, d_in[6] seq_offset: reproduced analytically.
  float* out = (float*)d_out;
  char*  ws  = (char*)d_ws;

  // workspace layout (~143 MB total)
  size_t oXB  = 0;                                    // x bf16      8 MB
  size_t oWQ  = oXB  + (size_t)Bc * Sc * Ec * 2;      // Wqkv bf16  24 MB
  size_t oWO  = oWQ  + (size_t)E3 * Ec * 2;           // Wout bf16   8 MB
  size_t oQKV = oWO  + (size_t)Ec * Ec * 2;           // qkv f32    48 MB
  size_t oQB  = oQKV + (size_t)Bc * Sc * E3 * 4;      // Q bf16      8 MB
  size_t oKB  = oQB  + (size_t)Bc * Hc * Sc * HDc * 2;    // K cache 16 MB
  size_t oVT  = oKB  + (size_t)Bc * Hc * LT * HDc * 2;    // V^T     16 MB
  size_t oAT  = oVT  + (size_t)Bc * Hc * HDc * LT * 2;    // attn out 8 MB

  bf16_t* xb   = (bf16_t*)(ws + oXB);
  bf16_t* wqb  = (bf16_t*)(ws + oWQ);
  bf16_t* wob  = (bf16_t*)(ws + oWO);
  float*  qkvf = (float*) (ws + oQKV);
  bf16_t* Qb   = (bf16_t*)(ws + oQB);
  bf16_t* Kb   = (bf16_t*)(ws + oKB);
  bf16_t* Vt   = (bf16_t*)(ws + oVT);
  bf16_t* attb = (bf16_t*)(ws + oAT);

  // 1) fp32 -> bf16 (x, Wqkv, Wout)
  {
    int nx = Bc * Sc * Ec / 8, nq = E3 * Ec / 8, no = Ec * Ec / 8;
    cvt_bf16_kernel<<<nx / 256, 256, 0, stream>>>(x,    xb,  nx);
    cvt_bf16_kernel<<<nq / 256, 256, 0, stream>>>(Wqkv, wqb, nq);
    cvt_bf16_kernel<<<no / 256, 256, 0, stream>>>(Wout, wob, no);
  }
  // 2) qkv = x @ Wqkv^T   (M=2048, N=6144, K=2048)
  gemm_bf16_kernel<<<dim3(E3 / 64, (Bc * Sc) / 128), 256, 0, stream>>>(
      xb, wqb, qkvf, Bc * Sc, E3, Ec);
  // 3) past K/V -> caches
  cache_copy_kernel<<<(Bc * SPc * Hc * HDc) / 256, 256, 0, stream>>>(pk, pv, Kb, Vt);
  // 4) RoPE(q,k) + scatter new k,v into caches
  rope_scatter_kernel<<<(Bc * Sc * Hc * HDc) / 256, 256, 0, stream>>>(qkvf, Qb, Kb, Vt);
  // 5) attention (flash, causal)
  attn_kernel<<<dim3(Bc * Hc, Sc / 64), 128, 0, stream>>>(Qb, Kb, Vt, attb);
  // 6) out = attn_out @ Wout^T  (M=2048, N=2048, K=2048), f32 result
  gemm_bf16_kernel<<<dim3(Ec / 64, (Bc * Sc) / 128), 256, 0, stream>>>(
      attb, wob, out, Bc * Sc, Ec, Ec);
}
